// Attention_40905268527506
// MI455X (gfx1250) — matrix operand order repaired and checked by reading, not compiled, not run on hardware
//
#include <hip/hip_runtime.h>
#include <hip/hip_bf16.h>

typedef __attribute__((ext_vector_type(16))) _Float16 v16h;
typedef __attribute__((ext_vector_type(8)))  float    v8f;
typedef __attribute__((ext_vector_type(4)))  float    v4f;

#define BATCH 64
#define SEQ   2048
#define DIM   512

// ---------------------------------------------------------------------------
// Kernel 1: scores[b,s] = sum_d x[b,s,d] * ctx[d]
// One wave computes 16 seq rows. A = 16x32 f16 tile of x, B = ctx replicated
// across all 16 columns, accumulate f32 in C over d. All D columns are equal;
// column 0 is written out. Scores are written into the weights region of
// d_out (overwritten in-place by the softmax kernel).
// ---------------------------------------------------------------------------
__global__ __launch_bounds__(256) void scores_kernel(const float* __restrict__ x,
                                                     const float* __restrict__ ctx,
                                                     float* __restrict__ scores) {
    const int lane   = threadIdx.x & 31;
    const int waveId = blockIdx.x * 8 + (threadIdx.x >> 5);   // 8192 waves total
    const int b      = waveId >> 7;                           // 128 waves per batch
    const int s0     = (waveId & 127) << 4;                   // 16 rows per wave
    const int row    = lane & 15;
    const int half   = lane >> 4;                             // 0 or 1
    const int kb     = half * 8;                              // A: K base per lane-half
    const int khalf  = half * 16;                             // unused; B uses kb like A

    const float* xr = x + ((size_t)b * SEQ + (size_t)(s0 + row)) * DIM;

    v8f c = {};
    #pragma unroll 4
    for (int d0 = 0; d0 < DIM; d0 += 32) {
        // A tile: lane holds K = kb..kb+7 and 16+kb..16+kb+7 of its row
        const float* pr = xr + d0 + kb;
        v4f f0 = *(const v4f*)(pr);
        v4f f1 = *(const v4f*)(pr + 4);
        v4f f2 = *(const v4f*)(pr + 16);
        v4f f3 = *(const v4f*)(pr + 20);
        v16h a;
        #pragma unroll
        for (int i = 0; i < 4; ++i) {
            a[i]      = (_Float16)f0[i];
            a[4 + i]  = (_Float16)f1[i];
            a[8 + i]  = (_Float16)f2[i];
            a[12 + i] = (_Float16)f3[i];
        }
        // B tile: B[k,n] = ctx[d0+k] for every column n (broadcast per half-wave)
        v16h bv;
        #pragma unroll
        for (int i = 0; i < 16; ++i) bv[i] = (_Float16)ctx[d0 + (i < 8 ? kb + i : 16 + kb + (i - 8))];

        c = __builtin_amdgcn_wmma_f32_16x16x32_f16(false, a, false, bv,
                                                   (short)0, c, false, false);
    }

    // Every column of D is identical; lanes with N==0 (lane 0 and 16) hold
    // D[0..7,0] and D[8..15,0] respectively.
    if ((lane & 15) == 0) {
        float* out = scores + (size_t)b * SEQ + s0 + half * 8;
        #pragma unroll
        for (int j = 0; j < 8; ++j) out[j] = c[j];
    }
}

// ---------------------------------------------------------------------------
// Kernel 2: in-place row softmax over SEQ=2048 elements. One block per batch.
// ---------------------------------------------------------------------------
__global__ __launch_bounds__(256) void softmax_kernel(float* __restrict__ w) {
    __shared__ float red[256];
    const int b = blockIdx.x;
    const int t = threadIdx.x;
    float* row = w + (size_t)b * SEQ;

    float v[8];
    float m = -3.402823466e38f;
    #pragma unroll
    for (int i = 0; i < 8; ++i) {
        v[i] = row[t + 256 * i];
        m = fmaxf(m, v[i]);
    }
    red[t] = m;
    __syncthreads();
    for (int s = 128; s > 0; s >>= 1) {
        if (t < s) red[t] = fmaxf(red[t], red[t + s]);
        __syncthreads();
    }
    m = red[0];
    __syncthreads();

    float sum = 0.0f;
    #pragma unroll
    for (int i = 0; i < 8; ++i) {
        v[i] = __expf(v[i] - m);
        sum += v[i];
    }
    red[t] = sum;
    __syncthreads();
    for (int s = 128; s > 0; s >>= 1) {
        if (t < s) red[t] += red[t + s];
        __syncthreads();
    }
    const float inv = 1.0f / red[0];
    #pragma unroll
    for (int i = 0; i < 8; ++i) row[t + 256 * i] = v[i] * inv;
}

// ---------------------------------------------------------------------------
// Kernel 3: pooled[b,d] = sum_s w[b,s] * x[b,s,d]
// One wave per (batch, 16-wide d tile). B = 32x16 f16 tile of x (column per
// lane; each per-lane load is a pair of 64B coalesced row segments across the
// wave), A = weights replicated across M. Accumulate C over 64 s-steps; every
// row of D equals pooled[d0..d0+15].
// ---------------------------------------------------------------------------
__global__ __launch_bounds__(256) void pool_kernel(const float* __restrict__ x,
                                                   const float* __restrict__ w,
                                                   float* __restrict__ pooled) {
    const int lane   = threadIdx.x & 31;
    const int waveId = blockIdx.x * 8 + (threadIdx.x >> 5);   // 2048 waves total
    const int b      = waveId >> 5;                           // 32 d-tiles per batch
    const int d0     = (waveId & 31) << 4;
    const int n      = lane & 15;
    const int half   = lane >> 4;
    const int kb     = half * 8;
    const int khalf  = half * 16;

    const float* xb = x + (size_t)b * SEQ * DIM;
    const float* wb = w + (size_t)b * SEQ;

    v8f c = {};
    for (int s0 = 0; s0 < SEQ; s0 += 32) {
        // B tile: B[k,n] = x[b, s0+k, d0+n]; this lane covers k = khalf..khalf+15
        const float* pc = xb + (size_t)(s0 + khalf) * DIM + d0 + n;
        v16h bv;
        #pragma unroll
        for (int i = 0; i < 16; ++i) bv[i] = (_Float16)pc[(size_t)i * DIM];

        // A tile: A[m,k] = w[b, s0+k] for all m (values depend only on k,
        // so per-lane-half broadcast yields the replicated matrix)
        const float* wr = wb + s0;
        v16h av;
        #pragma unroll
        for (int i = 0; i < 8; ++i) {
            av[i]     = (_Float16)wr[khalf + i];
            av[8 + i] = (_Float16)wr[khalf + 8 + i];
        }

        c = __builtin_amdgcn_wmma_f32_16x16x32_f16(false, av, false, bv,
                                                   (short)0, c, false, false);
    }

    // All D rows equal; lanes 0..15 hold D[0, n] in c[0].
    if (lane < 16) pooled[(size_t)b * DIM + d0 + n] = c[0];
}

// ---------------------------------------------------------------------------
extern "C" void kernel_launch(void* const* d_in, const int* in_sizes, int n_in,
                              void* d_out, int out_size, void* d_ws, size_t ws_size,
                              hipStream_t stream) {
    (void)in_sizes; (void)n_in; (void)out_size; (void)d_ws; (void)ws_size;

    const float* x   = (const float*)d_in[0];   // [64, 2048, 512] f32
    const float* ctx = (const float*)d_in[1];   // [512, 1] f32

    float* pooled  = (float*)d_out;                     // [64, 512]
    float* weights = (float*)d_out + BATCH * DIM;       // [64, 2048]

    // Pass 1: scores -> weights region of d_out (raw, pre-softmax)
    scores_kernel<<<1024, 256, 0, stream>>>(x, ctx, weights);
    // Pass 2: softmax in place
    softmax_kernel<<<BATCH, 256, 0, stream>>>(weights);
    // Pass 3: weighted pooling
    pool_kernel<<<256, 256, 0, stream>>>(x, weights, pooled);
}
